// MegatronQwenAttention_24575802868120
// MI455X (gfx1250) — compile-verified, matrix operand order checked by reading, not run on hardware
//
#include <hip/hip_runtime.h>
#include <hip/hip_bf16.h>
#include <math.h>

typedef __attribute__((ext_vector_type(16))) _Float16 v16h;
typedef __attribute__((ext_vector_type(8)))  _Float16 v8h;
typedef __attribute__((ext_vector_type(8)))  float    v8f;

#define SEQL    2048
#define HID     3584
#define NHEADS  28
#define NKV     4
#define HDIM    128
#define KVDIM   512
#define GQAREP  7              // 28 / 4
#define SCALE   0.08838834764831845f   // 1/sqrt(128)

// ---------------------------------------------------------------------------
// elementwise casts
// ---------------------------------------------------------------------------
__global__ void cast_f32_to_f16(const float* __restrict__ src,
                                _Float16* __restrict__ dst, size_t n) {
    size_t i = (size_t)blockIdx.x * blockDim.x + threadIdx.x;
    if (i < n) dst[i] = (_Float16)src[i];
}

__global__ void cast_copy_f16(const float* __restrict__ src,
                              _Float16* __restrict__ dstH,
                              float* __restrict__ dstF, size_t n) {
    size_t i = (size_t)blockIdx.x * blockDim.x + threadIdx.x;
    if (i < n) {
        float v = src[i];
        dstH[i] = (_Float16)v;
        dstF[i] = v;
    }
}

// src: K x N fp32 (row-major), dst: N x K f16 (row-major) == transpose+cast
__global__ void transpose_cast_f16(const float* __restrict__ src,
                                   _Float16* __restrict__ dst,
                                   int K, int N) {
    size_t i = (size_t)blockIdx.x * blockDim.x + threadIdx.x;
    size_t total = (size_t)K * N;
    if (i >= total) return;
    size_t n = i / (size_t)K;
    size_t k = i % (size_t)K;
    dst[n * (size_t)K + k] = (_Float16)src[k * (size_t)N + n];
}

// ---------------------------------------------------------------------------
// helper: build a v16h A/B fragment from two 16-byte chunks
// ---------------------------------------------------------------------------
__device__ __forceinline__ v16h frag32(const _Float16* base, int kbase) {
    v8h x0 = *(const v8h*)(base + kbase);
    v8h x1 = *(const v8h*)(base + kbase + 16);
    return __builtin_shufflevector(x0, x1, 0,1,2,3,4,5,6,7,8,9,10,11,12,13,14,15);
}

// ---------------------------------------------------------------------------
// WMMA GEMM:  C[M x N] (f32) = A[M x K] (f16, row-major) * Bt[N x K]^T + bias
// 32x32 register-blocked tile per wave: 2 A frags x 2 B frags -> 4 WMMA per
// K-step (halves fragment traffic per FLOP, 4 independent WMMAs for overlap).
// 8 waves / 256-thread block.
// ---------------------------------------------------------------------------
__global__ __launch_bounds__(256)
void gemm_wmma_f16(const _Float16* __restrict__ A,
                   const _Float16* __restrict__ Bt,
                   const float* __restrict__ bias,
                   float* __restrict__ C,
                   int M, int N, int K) {
    const int wave  = blockIdx.x * (blockDim.x >> 5) + (threadIdx.x >> 5);
    const int lane  = threadIdx.x & 31;
    const int lo    = lane & 15;
    const int hi    = lane >> 4;
    const int ntile = N >> 5;              // 32-wide N tiles
    const int tm = wave / ntile;
    const int tn = wave % ntile;
    if (tm >= (M >> 5)) return;

    const int kbase = hi * 8;
    const _Float16* arow0 = A  + (size_t)(tm * 32 + lo) * K;
    const _Float16* arow1 = A  + (size_t)(tm * 32 + 16 + lo) * K;
    const _Float16* brow0 = Bt + (size_t)(tn * 32 + lo) * K;
    const _Float16* brow1 = Bt + (size_t)(tn * 32 + 16 + lo) * K;

    v8f acc00 = {}, acc01 = {}, acc10 = {}, acc11 = {};
    for (int k0 = 0; k0 < K; k0 += 32) {
        if (k0 + 128 < K) {
            __builtin_prefetch(arow0 + k0 + 128, 0, 0);
            __builtin_prefetch(arow1 + k0 + 128, 0, 0);
            __builtin_prefetch(brow0 + k0 + 128, 0, 0);
            __builtin_prefetch(brow1 + k0 + 128, 0, 0);
        }
        v16h a0 = frag32(arow0 + k0, kbase);
        v16h a1 = frag32(arow1 + k0, kbase);
        v16h b0 = frag32(brow0 + k0, kbase);
        v16h b1 = frag32(brow1 + k0, kbase);
        acc00 = __builtin_amdgcn_wmma_f32_16x16x32_f16(false, a0, false, b0,
                                                       (short)0, acc00, false, false);
        acc01 = __builtin_amdgcn_wmma_f32_16x16x32_f16(false, a0, false, b1,
                                                       (short)0, acc01, false, false);
        acc10 = __builtin_amdgcn_wmma_f32_16x16x32_f16(false, a1, false, b0,
                                                       (short)0, acc10, false, false);
        acc11 = __builtin_amdgcn_wmma_f32_16x16x32_f16(false, a1, false, b1,
                                                       (short)0, acc11, false, false);
    }

    const int cn0 = tn * 32 + lo;
    const int cn1 = tn * 32 + 16 + lo;
    const float badd0 = bias ? bias[cn0] : 0.0f;
    const float badd1 = bias ? bias[cn1] : 0.0f;
#pragma unroll
    for (int r = 0; r < 8; ++r) {
        int m0 = tm * 32 + r + 8 * hi;
        int m1 = m0 + 16;
        C[(size_t)m0 * N + cn0] = acc00[r] + badd0;
        C[(size_t)m0 * N + cn1] = acc01[r] + badd1;
        C[(size_t)m1 * N + cn0] = acc10[r] + badd0;
        C[(size_t)m1 * N + cn1] = acc11[r] + badd1;
    }
}

// ---------------------------------------------------------------------------
// RoPE: rotate fp32 src in (seq, nheads, 128) layout; emit f16 copy and
// optionally fp32 cache (for cache_k output).
// ---------------------------------------------------------------------------
__global__ void rope_kernel(const float* __restrict__ src,
                            _Float16* __restrict__ dstH,
                            float* __restrict__ cacheF,
                            const int* __restrict__ pos_ids,
                            int nHeads) {
    int idx = blockIdx.x * blockDim.x + threadIdx.x;
    int total = SEQL * nHeads * 64;
    if (idx >= total) return;
    int i  = idx % 64;
    int t  = idx / 64;
    int hh = t % nHeads;
    int s  = t / nHeads;

    float p    = (float)pos_ids[s];
    float freq = powf(1000000.0f, -((float)i) * (1.0f / 64.0f));
    float ang  = p * freq;
    float c = cosf(ang), sn = sinf(ang);

    size_t base = (size_t)s * (nHeads * HDIM) + (size_t)hh * HDIM;
    float x1 = src[base + i];
    float x2 = src[base + i + 64];
    float o1 = x1 * c - x2 * sn;
    float o2 = x1 * sn + x2 * c;

    dstH[base + i]      = (_Float16)o1;
    dstH[base + i + 64] = (_Float16)o2;
    if (cacheF) {
        cacheF[base + i]      = o1;
        cacheF[base + i + 64] = o2;
    }
}

// ---------------------------------------------------------------------------
// Flash attention: 1 wave per (16-query tile, head). Online softmax fp32,
// S = Q K^T and O += P V via WMMA. P transposed D->A layout via LDS,
// V tile staged in LDS with 128-bit copies.
// ---------------------------------------------------------------------------
__global__ __launch_bounds__(32)
void flash_attn_wmma(const _Float16* __restrict__ Q,   // SEQL x HID
                     const _Float16* __restrict__ Kc,  // SEQL x KVDIM
                     const _Float16* __restrict__ V,   // SEQL x KVDIM
                     const float* __restrict__ mask,   // SEQL x SEQL
                     float* __restrict__ O) {          // SEQL x HID (f32)
    __shared__ __align__(16) _Float16 Plds[16 * 32];
    __shared__ __align__(16) _Float16 Vlds[32 * HDIM];

    const int qt   = blockIdx.x;           // 0..127
    const int h    = blockIdx.y;           // 0..27
    const int kvh  = h / GQAREP;
    const int lane = threadIdx.x & 31;
    const int lo   = lane & 15;
    const int hi   = lane >> 4;
    const int kbase = hi * 8;

    // --- Q fragments: 16 rows x 128 dims = 4 K-chunks of 32 ---
    v16h aq[4];
    const _Float16* qrow = Q + (size_t)(qt * 16 + lo) * HID + h * HDIM;
#pragma unroll
    for (int c = 0; c < 4; ++c) aq[c] = frag32(qrow + c * 32, kbase);

    float mrow[8], lrow[8];
    v8f o[8];
    v8f zero = {};
#pragma unroll
    for (int r = 0; r < 8; ++r) { mrow[r] = -1e30f; lrow[r] = 0.0f; o[r] = zero; }

    const int kmax  = qt * 16 + 16;        // exclusive causal bound
    const int steps = (kmax + 31) >> 5;

    for (int stp = 0; stp < steps; ++stp) {
        const int kb = stp * 32;

        // stage V tile (32 keys x 128 dims) -> LDS, 128-bit copies
        {
            const v8h* vrow = (const v8h*)(V + (size_t)(kb + lane) * KVDIM + kvh * HDIM);
            v8h* dst = (v8h*)(Vlds + lane * HDIM);
#pragma unroll
            for (int c = 0; c < 16; ++c) dst[c] = vrow[c];
        }

        // --- S = Q K^T : two 16x16 tiles (keys kb..kb+15, kb+16..kb+31) ---
        v8f s0 = zero, s1 = zero;
#pragma unroll
        for (int c = 0; c < 4; ++c) {
            const _Float16* k0r = Kc + (size_t)(kb + lo) * KVDIM + kvh * HDIM + c * 32;
            v16h b0 = frag32(k0r, kbase);
            s0 = __builtin_amdgcn_wmma_f32_16x16x32_f16(false, aq[c], false, b0,
                                                        (short)0, s0, false, false);
            const _Float16* k1r = Kc + (size_t)(kb + 16 + lo) * KVDIM + kvh * HDIM + c * 32;
            v16h b1 = frag32(k1r, kbase);
            s1 = __builtin_amdgcn_wmma_f32_16x16x32_f16(false, aq[c], false, b1,
                                                        (short)0, s1, false, false);
        }

        // --- scale + mask + online softmax update; write P to LDS ---
#pragma unroll
        for (int r = 0; r < 8; ++r) {
            int qg = qt * 16 + r + 8 * hi;
            float e0 = s0[r] * SCALE + mask[(size_t)qg * SEQL + kb + lo];
            float e1 = s1[r] * SCALE + mask[(size_t)qg * SEQL + kb + 16 + lo];
            float mx = fmaxf(e0, e1);
            mx = fmaxf(mx, __shfl_xor(mx, 1, 32));
            mx = fmaxf(mx, __shfl_xor(mx, 2, 32));
            mx = fmaxf(mx, __shfl_xor(mx, 4, 32));
            mx = fmaxf(mx, __shfl_xor(mx, 8, 32));
            float mnew = fmaxf(mrow[r], mx);
            float fr = expf(mrow[r] - mnew);
            float p0 = expf(e0 - mnew);
            float p1 = expf(e1 - mnew);
            float sum = p0 + p1;
            sum += __shfl_xor(sum, 1, 32);
            sum += __shfl_xor(sum, 2, 32);
            sum += __shfl_xor(sum, 4, 32);
            sum += __shfl_xor(sum, 8, 32);
            lrow[r] = lrow[r] * fr + sum;
            mrow[r] = mnew;
#pragma unroll
            for (int dc = 0; dc < 8; ++dc) o[dc][r] *= fr;
            Plds[(r + 8 * hi) * 32 + lo]      = (_Float16)p0;
            Plds[(r + 8 * hi) * 32 + 16 + lo] = (_Float16)p1;
        }
        __syncthreads();

        // --- A fragment of P (16 x 32) from LDS ---
        v16h ap;
#pragma unroll
        for (int j = 0; j < 8; ++j) {
            ap[j]     = Plds[lo * 32 + kbase + j];
            ap[j + 8] = Plds[lo * 32 + kbase + 16 + j];
        }

        // --- O += P * V : 8 WMMA over the 8 dim-chunks of 16 ---
#pragma unroll
        for (int dc = 0; dc < 8; ++dc) {
            const int d = dc * 16 + lo;
            v16h bv;
#pragma unroll
            for (int j = 0; j < 8; ++j) {
                bv[j]     = Vlds[(kbase + j) * HDIM + d];
                bv[j + 8] = Vlds[(kbase + 16 + j) * HDIM + d];
            }
            o[dc] = __builtin_amdgcn_wmma_f32_16x16x32_f16(false, ap, false, bv,
                                                           (short)0, o[dc], false, false);
        }
        __syncthreads();
    }

    // --- epilogue: normalize and store fp32 ---
#pragma unroll
    for (int r = 0; r < 8; ++r) {
        float inv = 1.0f / lrow[r];
        int qg = qt * 16 + r + 8 * hi;
        float* orow = O + (size_t)qg * HID + h * HDIM;
#pragma unroll
        for (int dc = 0; dc < 8; ++dc) orow[dc * 16 + lo] = o[dc][r] * inv;
    }
}

// ---------------------------------------------------------------------------
// host launcher
// ---------------------------------------------------------------------------
extern "C" void kernel_launch(void* const* d_in, const int* in_sizes, int n_in,
                              void* d_out, int out_size, void* d_ws, size_t ws_size,
                              hipStream_t stream) {
    const float* hidden = (const float*)d_in[0];
    const float* maskp  = (const float*)d_in[1];
    const int*   posids = (const int*)d_in[2];
    const float* wq = (const float*)d_in[3];
    const float* bq = (const float*)d_in[4];
    const float* wk = (const float*)d_in[5];
    const float* bk = (const float*)d_in[6];
    const float* wv = (const float*)d_in[7];
    const float* bv = (const float*)d_in[8];
    const float* wo = (const float*)d_in[9];

    float* out    = (float*)d_out;                         // SEQL*HID
    float* cacheK = out + (size_t)SEQL * HID;              // SEQL*KVDIM
    float* cacheV = cacheK + (size_t)SEQL * KVDIM;         // SEQL*KVDIM

    const size_t nHS = (size_t)SEQL * HID;                 // 7340032
    const size_t nKVe = (size_t)SEQL * KVDIM;              // 1048576

    // workspace carve (256B aligned chunks)
    char* p = (char*)d_ws;
    auto carve = [&](size_t bytes) -> void* {
        void* r = (void*)p;
        p += (bytes + 255) & ~(size_t)255;
        return r;
    };
    _Float16* hsH  = (_Float16*)carve(nHS * 2);
    _Float16* wqT  = (_Float16*)carve((size_t)HID * HID * 2);
    _Float16* wkT  = (_Float16*)carve((size_t)HID * KVDIM * 2);
    _Float16* wvT  = (_Float16*)carve((size_t)HID * KVDIM * 2);
    _Float16* woT  = (_Float16*)carve((size_t)HID * HID * 2);
    float*    qF   = (float*)carve(nHS * 4);
    float*    kF   = (float*)carve(nKVe * 4);
    float*    vF   = (float*)carve(nKVe * 4);
    _Float16* qH   = (_Float16*)carve(nHS * 2);
    _Float16* kH   = (_Float16*)carve(nKVe * 2);
    _Float16* vH   = (_Float16*)carve(nKVe * 2);
    // reuse dead buffers: attnF aliases qF (q fp32 dead after RoPE),
    // attnH aliases hsH (hidden f16 dead after QKV GEMMs)
    float*    attnF = qF;
    _Float16* attnH = hsH;

    const int THR = 256;

    // 1) casts / transposes
    cast_f32_to_f16<<<(unsigned)((nHS + THR - 1) / THR), THR, 0, stream>>>(hidden, hsH, nHS);
    {
        size_t t = (size_t)HID * HID;
        transpose_cast_f16<<<(unsigned)((t + THR - 1) / THR), THR, 0, stream>>>(wq, wqT, HID, HID);
        transpose_cast_f16<<<(unsigned)((t + THR - 1) / THR), THR, 0, stream>>>(wo, woT, HID, HID);
        size_t t2 = (size_t)HID * KVDIM;
        transpose_cast_f16<<<(unsigned)((t2 + THR - 1) / THR), THR, 0, stream>>>(wk, wkT, HID, KVDIM);
        transpose_cast_f16<<<(unsigned)((t2 + THR - 1) / THR), THR, 0, stream>>>(wv, wvT, HID, KVDIM);
    }

    // 2) QKV projections (WMMA, 32x32 tiles per wave)
    {
        int wavesQ = (SEQL / 32) * (HID / 32);    // 64 * 112 = 7168
        int wavesK = (SEQL / 32) * (KVDIM / 32);  // 64 * 16  = 1024
        gemm_wmma_f16<<<wavesQ / 8, 256, 0, stream>>>(hsH, wqT, bq, qF, SEQL, HID, HID);
        gemm_wmma_f16<<<wavesK / 8, 256, 0, stream>>>(hsH, wkT, bk, kF, SEQL, KVDIM, HID);
        gemm_wmma_f16<<<wavesK / 8, 256, 0, stream>>>(hsH, wvT, bv, vF, SEQL, KVDIM, HID);
    }

    // 3) RoPE (q, k) + V cast / caches
    {
        int nq = SEQL * NHEADS * 64;
        int nk = SEQL * NKV * 64;
        rope_kernel<<<(nq + THR - 1) / THR, THR, 0, stream>>>(qF, qH, nullptr, posids, NHEADS);
        rope_kernel<<<(nk + THR - 1) / THR, THR, 0, stream>>>(kF, kH, cacheK, posids, NKV);
        cast_copy_f16<<<(unsigned)((nKVe + THR - 1) / THR), THR, 0, stream>>>(vF, vH, cacheV, nKVe);
    }

    // 4) flash attention (WMMA), writes attnF (aliases qF)
    flash_attn_wmma<<<dim3(SEQL / 16, NHEADS), 32, 0, stream>>>(qH, kH, vH, maskp, attnF);

    // 5) output projection
    cast_f32_to_f16<<<(unsigned)((nHS + THR - 1) / THR), THR, 0, stream>>>(attnF, attnH, nHS);
    {
        int wavesO = (SEQL / 32) * (HID / 32);
        gemm_wmma_f16<<<wavesO / 8, 256, 0, stream>>>(attnH, woT, nullptr, out, SEQL, HID, HID);
    }
}
